// CausalSelfAttention_40853728919770
// MI455X (gfx1250) — matrix operand order repaired and checked by reading, not compiled, not run on hardware
//
#include <hip/hip_runtime.h>

// ---------------------------------------------------------------------------
// CausalSelfAttention forward for MI455X (gfx1250, wave32, WMMA).
// Pipeline: f32->bf16 convert; QKV GEMM (WMMA bf16, async-to-LDS A staging,
// fused head-transpose scatter); per-wave flash attention (WMMA bf16, online
// softmax, f32 state); output-projection GEMM (WMMA bf16 -> f32).
// ---------------------------------------------------------------------------

typedef __bf16 bf16;
typedef __bf16 bf16x16 __attribute__((ext_vector_type(16)));
typedef float  f32x8   __attribute__((ext_vector_type(8)));

#define N_EMBD   1024
#define N_HEAD   16
#define HEAD_DIM 64
#define BATCH    4
#define SEQ      2048
#define TOKENS   (BATCH * SEQ)   // 8192

// CDNA5 WMMA 16x16x32 bf16 fragment index helpers (per CDNA5 ISA):
// A (16x32): lane row = lane&15; VGPR p holds K pair at:
//   p<4 : K = hi*8 + 2p ; p>=4: K = 16 + hi*8 + 2(p-4)
__device__ __forceinline__ int kbA(int p, int hi) {
    return ((p & 4) << 2) + hi * 8 + ((p & 3) << 1);
}
// B (32x16): lane col = lane&15; contiguous K pair {hi*16 + 2p, +1}, not the ISA order; unused, B frags use kbA
__device__ __forceinline__ int kbB(int p, int hi) {
    return hi * 16 + (p << 1);
}

union Frag16 { bf16x16 v; unsigned int u[8]; };
union PackBf { unsigned int u; bf16 h[2]; };

// LDS byte "offset" for async-to-LDS VDST operand. Flat shared addresses put
// the LDS offset in the low 32 bits (aperture in the high bits), and HW
// truncates to LDS space, so the low dword is the right VGPR payload.
__device__ __forceinline__ unsigned lds_addr32(const void* p) {
    return (unsigned)(size_t)p;
}

// ---------------------------------------------------------------------------
// Elementwise f32 -> bf16 conversion
// ---------------------------------------------------------------------------
__global__ void f32_to_bf16_kernel(const float* __restrict__ src,
                                   bf16* __restrict__ dst, int n) {
    int i = blockIdx.x * blockDim.x + threadIdx.x;
    if (i < n) dst[i] = (bf16)src[i];
}

// ---------------------------------------------------------------------------
// Tiled bf16 WMMA GEMM: C[M,N] = A[M,K] @ B[K,N] + bias[N]
//   mode 0: scatter qkv output as bf16 into Q/K/V with [B,H,T,D] layout
//   mode 1: store f32 into outF[M,N]
// Block: 128 threads = 4 waves. Tile: 128(M) x 64(N), K-step 32.
// A tile staged with GLOBAL_LOAD_ASYNC_TO_LDS_B128 (ASYNCcnt path); B tile
// staged manually because it is transposed into LDS. LDS rows padded to
// 34 bf16 (17 dwords) -> conflict-free fragment gathers.
// ---------------------------------------------------------------------------
#define TM 128
#define TN 64
#define TK 32

__global__ __launch_bounds__(128) void gemm_bf16_wmma(
    const bf16* __restrict__ A, const bf16* __restrict__ Bm,
    const float* __restrict__ bias,
    int M, int N, int K, int mode,
    float* __restrict__ outF,
    bf16* __restrict__ Qb, bf16* __restrict__ Kb, bf16* __restrict__ Vb)
{
    __shared__ bf16 As[TM * 34];   // [128][32] + pad
    __shared__ bf16 Bs[TN * 34];   // transposed: [n=64][k=32] + pad

    const int tid  = threadIdx.x;
    const int lane = tid & 31;
    const int wave = tid >> 5;       // 0..3 -> 32-row slab
    const int ln   = lane & 15;
    const int hi   = lane >> 4;
    const int m0   = blockIdx.y * TM;
    const int n0   = blockIdx.x * TN;

    // ---- hoisted per-thread staging addresses ----
    // A: 128x32 bf16 tile = 512 x b128 chunks; 4 chunks per thread.
    unsigned   ldsA[4];
    const bf16* gA[4];
    for (int j = 0; j < 4; ++j) {
        int idx = tid + j * 128;     // chunk id
        int r   = idx >> 2;          // row 0..127
        int sg  = (idx & 3) << 3;    // 8-elem segment within row
        ldsA[j] = lds_addr32(&As[r * 34 + sg]);
        gA[j]   = A + (size_t)(m0 + r) * K + sg;
    }
    // B: 32x64 tile as dwords; 8 dwords per thread, transposed store offsets.
    int gBoff[8], sBoff[8];
    for (int i = 0; i < 8; ++i) {
        int idx  = tid + i * 128;
        int kk   = idx >> 5;             // 0..31
        int n2   = (idx & 31) << 1;      // 0..62
        gBoff[i] = kk * N + n2;          // + k0*N + n0 at use
        sBoff[i] = n2 * 34 + kk;         // row n2 (and +34 for n2+1)
    }

    f32x8 acc[2][4];
    for (int a = 0; a < 2; ++a)
        for (int b = 0; b < 4; ++b)
            for (int p = 0; p < 8; ++p) acc[a][b][p] = 0.0f;

    for (int k0 = 0; k0 < K; k0 += TK) {
        // Async-stage A tile straight into LDS (no VGPR round trip).
        for (int j = 0; j < 4; ++j) {
            asm volatile("global_load_async_to_lds_b128 %0, %1, off"
                         :: "v"(ldsA[j]), "v"(gA[j]) : "memory");
            gA[j] += TK;
        }
        // Prefetch next A tile into cache while this one streams.
        if (k0 + TK < K) __builtin_prefetch(gA[0], 0, 0);

        // Stage B tile 32x64, transposed into Bs[n][k] so K-pairs are dwords.
        {
            const bf16* Bk = Bm + (size_t)k0 * N + n0;
            for (int i = 0; i < 8; ++i) {
                PackBf cv;
                cv.u = *(const unsigned int*)&Bk[gBoff[i]];
                Bs[sBoff[i]]      = cv.h[0];
                Bs[sBoff[i] + 34] = cv.h[1];
            }
        }

        asm volatile("s_wait_asynccnt 0x0" ::: "memory");
        __syncthreads();

        Frag16 af[2];
        for (int am = 0; am < 2; ++am) {
            int r = wave * 32 + am * 16 + ln;
            for (int p = 0; p < 8; ++p)
                af[am].u[p] = *(const unsigned int*)&As[r * 34 + kbA(p, hi)];
        }
        for (int bf = 0; bf < 4; ++bf) {
            Frag16 bb;
            int n = bf * 16 + ln;
            for (int p = 0; p < 8; ++p)
                bb.u[p] = *(const unsigned int*)&Bs[n * 34 + kbA(p, hi)];
            for (int am = 0; am < 2; ++am)
                acc[am][bf] = __builtin_amdgcn_wmma_f32_16x16x32_bf16(
                    false, af[am].v, false, bb.v, (short)0, acc[am][bf],
                    false, false);
        }
        __syncthreads();
    }

    // Epilogue: bias add + store.
    for (int am = 0; am < 2; ++am) {
        for (int bf = 0; bf < 4; ++bf) {
            int n_g  = n0 + bf * 16 + ln;
            float bv = bias[n_g];
            for (int p = 0; p < 8; ++p) {
                int m_g = m0 + wave * 32 + am * 16 + p + 8 * hi;
                float v = acc[am][bf][p] + bv;
                if (mode == 0) {
                    // qkv scatter: n in [0,3072): part|head|dim; m: batch|token
                    int part = n_g >> 10;
                    int c = n_g & (N_EMBD - 1);
                    int h = c >> 6, d = c & (HEAD_DIM - 1);
                    int b = m_g >> 11, t = m_g & (SEQ - 1);
                    bf16* dst = (part == 0) ? Qb : (part == 1) ? Kb : Vb;
                    dst[((size_t)(b * N_HEAD + h) * SEQ + t) * HEAD_DIM + d] =
                        (bf16)v;
                } else {
                    outF[(size_t)m_g * N + n_g] = v;
                }
            }
        }
    }
}

// ---------------------------------------------------------------------------
// Flash attention: one wave32 per (b,h, 16-query-row tile).
// Per 32-key chunk: 4 WMMAs (S = Q K^T), online softmax (shfl_xor row
// reductions inside 16-lane halves), P relayout via LDS, 4 WMMAs (O += P V).
// f32 running max/sum/accumulator; bf16 operands.
// ---------------------------------------------------------------------------
__global__ __launch_bounds__(32) void attn_fwd(
    const bf16* __restrict__ Q, const bf16* __restrict__ K,
    const bf16* __restrict__ V, bf16* __restrict__ Y)
{
    __shared__ bf16 Ps[16 * 34];

    const int lane = threadIdx.x;
    const int ln   = lane & 15;
    const int hi   = lane >> 4;
    const int qt   = blockIdx.x & (SEQ / 16 - 1);  // 128 q-tiles
    const int bh   = blockIdx.x >> 7;
    const int q0   = qt * 16;
    const size_t baseT = (size_t)bh * SEQ;

    // Q fragments (rows q0..q0+15, d split into two 32-wide K chunks).
    Frag16 qf[2];
    {
        const bf16* qrow = Q + (baseT + q0 + ln) * HEAD_DIM;
        for (int dc = 0; dc < 2; ++dc)
            for (int p = 0; p < 8; ++p)
                qf[dc].u[p] =
                    *(const unsigned int*)(qrow + dc * 32 + kbA(p, hi));
    }

    float mrow[8], lrow[8];
    f32x8 O[4];
    for (int p = 0; p < 8; ++p) { mrow[p] = -1e30f; lrow[p] = 0.0f; }
    for (int dc = 0; dc < 4; ++dc)
        for (int p = 0; p < 8; ++p) O[dc][p] = 0.0f;

    const float scale = 0.125f;  // 1/sqrt(64)

    for (int k0 = 0; k0 < q0 + 16; k0 += 32) {
        // Prefetch next key chunk while computing this one.
        if (k0 + 32 < q0 + 16) {
            __builtin_prefetch(K + (baseT + k0 + 32 + lane) * HEAD_DIM, 0, 0);
            __builtin_prefetch(V + (baseT + k0 + 32 + lane) * HEAD_DIM, 0, 0);
        }

        // ---- S = Q K^T for 32 keys (two 16-wide key halves) ----
        f32x8 S[2];
        for (int kh = 0; kh < 2; ++kh) {
            for (int p = 0; p < 8; ++p) S[kh][p] = 0.0f;
            int key = k0 + kh * 16 + ln;
            const bf16* krow = K + (baseT + key) * HEAD_DIM;
            for (int dc = 0; dc < 2; ++dc) {
                Frag16 kb;  // B-frag of K^T: K-dim = d, N = key
                for (int p = 0; p < 8; ++p)
                    kb.u[p] =
                        *(const unsigned int*)(krow + dc * 32 + kbA(p, hi));
                S[kh] = __builtin_amdgcn_wmma_f32_16x16x32_bf16(
                    false, qf[dc].v, false, kb.v, (short)0, S[kh],
                    false, false);
            }
        }

        // ---- causal mask, online softmax update ----
        float pr[2][8];
        for (int p = 0; p < 8; ++p) {
            int row = q0 + p + 8 * hi;
            float s0 = S[0][p] * scale; if (k0 + ln > row)      s0 = -1e30f;
            float s1 = S[1][p] * scale; if (k0 + 16 + ln > row) s1 = -1e30f;
            float mx = fmaxf(s0, s1);
            mx = fmaxf(mx, __shfl_xor(mx, 1));
            mx = fmaxf(mx, __shfl_xor(mx, 2));
            mx = fmaxf(mx, __shfl_xor(mx, 4));
            mx = fmaxf(mx, __shfl_xor(mx, 8));
            float mnew  = fmaxf(mrow[p], mx);
            float alpha = __expf(mrow[p] - mnew);
            float p0 = __expf(s0 - mnew);
            float p1 = __expf(s1 - mnew);
            float rs = p0 + p1;
            rs += __shfl_xor(rs, 1);
            rs += __shfl_xor(rs, 2);
            rs += __shfl_xor(rs, 4);
            rs += __shfl_xor(rs, 8);
            lrow[p] = lrow[p] * alpha + rs;
            mrow[p] = mnew;
            for (int dc = 0; dc < 4; ++dc) O[dc][p] *= alpha;
            pr[0][p] = p0;
            pr[1][p] = p1;
        }

        // ---- P: C-fragment layout -> A-fragment layout via LDS ----
        for (int kh = 0; kh < 2; ++kh)
            for (int p = 0; p < 8; ++p)
                Ps[(p + 8 * hi) * 34 + kh * 16 + ln] = (bf16)pr[kh][p];
        __syncthreads();  // single-wave WG: barrier is NOP, fence orders LDS
        Frag16 pf;
        for (int p = 0; p < 8; ++p)
            pf.u[p] = *(const unsigned int*)&Ps[ln * 34 + kbA(p, hi)];
        __syncthreads();

        // ---- O += P @ V ----
        const bf16* vbase = V + baseT * HEAD_DIM;
        for (int dc = 0; dc < 4; ++dc) {
            Frag16 vf;  // B-frag of V: K-dim = key(32), N = d(16)
            for (int p = 0; p < 8; ++p) {
                int kk = kbA(p, hi);
                PackBf cv;
                cv.h[0] = vbase[(size_t)(k0 + kk + 0) * HEAD_DIM + dc * 16 + ln];
                cv.h[1] = vbase[(size_t)(k0 + kk + 1) * HEAD_DIM + dc * 16 + ln];
                vf.u[p] = cv.u;
            }
            O[dc] = __builtin_amdgcn_wmma_f32_16x16x32_bf16(
                false, pf.v, false, vf.v, (short)0, O[dc], false, false);
        }
    }

    // ---- normalize and store to [B,T,C] bf16 (un-transpose heads) ----
    const int b = bh >> 4, h = bh & (N_HEAD - 1);
    for (int p = 0; p < 8; ++p) {
        float inv = 1.0f / lrow[p];
        int t = q0 + p + 8 * hi;
        size_t rowbase = ((size_t)(b * SEQ + t)) * N_EMBD + h * HEAD_DIM;
        for (int dc = 0; dc < 4; ++dc)
            Y[rowbase + dc * 16 + ln] = (bf16)(O[dc][p] * inv);
    }
}

// ---------------------------------------------------------------------------
// Launch
// ---------------------------------------------------------------------------
extern "C" void kernel_launch(void* const* d_in, const int* in_sizes, int n_in,
                              void* d_out, int out_size, void* d_ws,
                              size_t ws_size, hipStream_t stream)
{
    const float* x      = (const float*)d_in[0];
    const float* W_qkv  = (const float*)d_in[1];
    const float* b_qkv  = (const float*)d_in[2];
    const float* W_proj = (const float*)d_in[3];
    const float* b_proj = (const float*)d_in[4];
    float* out = (float*)d_out;

    // Workspace layout (88 MB total, bf16 staging):
    char* ws = (char*)d_ws;
    size_t off = 0;
    bf16* xb    = (bf16*)(ws + off); off += (size_t)TOKENS * N_EMBD * 2;      // 16 MB
    bf16* wqkvb = (bf16*)(ws + off); off += (size_t)N_EMBD * 3 * N_EMBD * 2;  //  6 MB
    bf16* wprjb = (bf16*)(ws + off); off += (size_t)N_EMBD * N_EMBD * 2;      //  2 MB
    bf16* Qb    = (bf16*)(ws + off); off += (size_t)TOKENS * N_EMBD * 2;      // 16 MB
    bf16* Kb    = (bf16*)(ws + off); off += (size_t)TOKENS * N_EMBD * 2;      // 16 MB
    bf16* Vb    = (bf16*)(ws + off); off += (size_t)TOKENS * N_EMBD * 2;      // 16 MB
    bf16* Yb    = (bf16*)(ws + off); off += (size_t)TOKENS * N_EMBD * 2;      // 16 MB
    (void)ws_size; (void)in_sizes; (void)n_in; (void)out_size;

    // 1) convert inputs to bf16
    {
        int n = TOKENS * N_EMBD;
        f32_to_bf16_kernel<<<(n + 255) / 256, 256, 0, stream>>>(x, xb, n);
        n = N_EMBD * 3 * N_EMBD;
        f32_to_bf16_kernel<<<(n + 255) / 256, 256, 0, stream>>>(W_qkv, wqkvb, n);
        n = N_EMBD * N_EMBD;
        f32_to_bf16_kernel<<<(n + 255) / 256, 256, 0, stream>>>(W_proj, wprjb, n);
    }

    // 2) qkv = x @ W_qkv + b_qkv, scattered to Q/K/V [B,H,T,D] bf16
    {
        dim3 grid(3 * N_EMBD / TN, TOKENS / TM);
        gemm_bf16_wmma<<<grid, 128, 0, stream>>>(
            xb, wqkvb, b_qkv, TOKENS, 3 * N_EMBD, N_EMBD, /*mode=*/0,
            nullptr, Qb, Kb, Vb);
    }

    // 3) flash attention, one wave per 16 query rows
    attn_fwd<<<BATCH * N_HEAD * (SEQ / 16), 32, 0, stream>>>(Qb, Kb, Vb, Yb);

    // 4) y = attn @ W_proj + b_proj -> f32 out
    {
        dim3 grid(N_EMBD / TN, TOKENS / TM);
        gemm_bf16_wmma<<<grid, 128, 0, stream>>>(
            Yb, wprjb, b_proj, TOKENS, N_EMBD, N_EMBD, /*mode=*/1,
            out, nullptr, nullptr, nullptr);
    }
}
